// Decoder_69415261438397
// MI455X (gfx1250) — matrix operand order repaired and checked by reading, not compiled, not run on hardware
//
#include <hip/hip_runtime.h>
#include <hip/hip_bf16.h>
#include <math.h>

#define B_   128
#define T_   512
#define E_   128
#define D_   128

typedef _Float16 v16h __attribute__((ext_vector_type(16)));
typedef _Float16 v8h  __attribute__((ext_vector_type(8)));
typedef float    v8f  __attribute__((ext_vector_type(8)));

// ---------------- workspace layout (bytes, 256B aligned chunks) ----------------
static constexpr size_t OFF_UENC  = 0;
static constexpr size_t SZ_UENC   = (size_t)B_ * T_ * T_ * sizeof(_Float16); // 67 MB, L2-resident
static constexpr size_t OFF_S     = OFF_UENC  + SZ_UENC;
static constexpr size_t OFF_SC    = OFF_S     + (size_t)B_ * T_ * 4;
static constexpr size_t OFF_Z     = OFF_SC    + (size_t)B_ * T_ * 4;
static constexpr size_t OFF_H     = OFF_Z     + (size_t)B_ * 4 * D_ * 4;
static constexpr size_t OFF_C     = OFF_H     + (size_t)B_ * D_ * 4;
static constexpr size_t OFF_CTX   = OFF_C     + (size_t)B_ * D_ * 4;
static constexpr size_t OFF_SYNC  = OFF_CTX   + (size_t)B_ * E_ * 4;
static constexpr size_t OFF_ENC16 = OFF_SYNC  + 256;                          // f16 copy of enc
static constexpr size_t OFF_WAT   = OFF_ENC16 + (size_t)B_ * T_ * E_ * 2;     // Wa^T  (T x 2D) f16
static constexpr size_t OFF_WRT   = OFF_WAT   + (size_t)T_ * 2 * D_ * 2;      // Wr^T  (4D x D) f16
static constexpr size_t OFF_UAT   = OFF_WRT   + (size_t)4 * D_ * D_ * 2;      // Ua^T  (T x E)  f16
static constexpr size_t OFF_HC16  = OFF_UAT   + (size_t)T_ * E_ * 2;          // [h|c] (B x 2D) f16

// ---------------- helpers ----------------
// Hardware transcendental tanh (gfx1250 V_TANH_F32). ISA TRANS hazard: one
// independent op / V_NOP before the result is consumed -> emit v_nop inline.
__device__ __forceinline__ float htanh(float x) {
  float r;
  asm volatile("v_tanh_f32 %0, %1\n\tv_nop" : "=v"(r) : "v"(x));
  return r;
}
__device__ __forceinline__ float sigf(float x) { return 0.5f * htanh(0.5f * x) + 0.5f; }

// Device-wide barrier (persistent-kernel scan). One arrival per block.
__device__ __forceinline__ void grid_sync(unsigned* cnt, unsigned* gen) {
  __syncthreads();
  if (threadIdx.x == 0) {
    __threadfence(); // DEV-scope release of this block's writes
    unsigned g = __hip_atomic_load(gen, __ATOMIC_ACQUIRE, __HIP_MEMORY_SCOPE_AGENT);
    unsigned arrived =
        __hip_atomic_fetch_add(cnt, 1u, __ATOMIC_ACQ_REL, __HIP_MEMORY_SCOPE_AGENT) + 1u;
    if (arrived == gridDim.x) {
      __hip_atomic_store(cnt, 0u, __ATOMIC_RELAXED, __HIP_MEMORY_SCOPE_AGENT);
      __hip_atomic_fetch_add(gen, 1u, __ATOMIC_RELEASE, __HIP_MEMORY_SCOPE_AGENT);
    } else {
      while (__hip_atomic_load(gen, __ATOMIC_ACQUIRE, __HIP_MEMORY_SCOPE_AGENT) == g)
        __builtin_amdgcn_s_sleep(1);
    }
  }
  __syncthreads();
}

// One 16x16 f32 tile: A (16xK f16, row-major, lda) x B (Kx16, given as BT = K-major
// per column so each lane's 16 K-values are contiguous -> 2x b128 loads each side).
// ISA VGPR layouts (CDNA5 ISA):
//   A 16x32 f16 : lane L -> M=L%16; elem v*2+h -> K=(v&3)*2+h + (L/16)*8 + (v/4)*16
//     => a[0..7] = A[m][k0+kh*8 .. +7],  a[8..15] = A[m][k0+16+kh*8 .. +7]
//   B 32x16 f16 : lane L -> N=L%16; same K order as A
//     => b[0..7] = BT[n][k0+kh*8 .. +7],  b[8..15] = BT[n][k0+16+kh*8 .. +7]
//   C 16x16 f32 : vgpr r lane L -> M=r+8*(L/16), N=L%16
__device__ __forceinline__ v8f wmma_tile(v8f acc, const _Float16* __restrict__ Ab, int lda,
                                         const _Float16* __restrict__ BTb, int ldk,
                                         int K, int lane) {
  const int mlo = lane & 15;
  const int kh  = lane >> 4;
  const _Float16* ap = Ab  + (size_t)mlo * lda + (kh << 3);
  const _Float16* bp = BTb + (size_t)mlo * ldk + (kh << 3);
  for (int k0 = 0; k0 < K; k0 += 32) {
    v8h a0 = *(const v8h*)(ap + k0);
    v8h a1 = *(const v8h*)(ap + k0 + 16);
    v8h b0 = *(const v8h*)(bp + k0);
    v8h b1 = *(const v8h*)(bp + k0 + 16);
    v16h a = __builtin_shufflevector(a0, a1, 0, 1, 2, 3, 4, 5, 6, 7,
                                             8, 9, 10, 11, 12, 13, 14, 15);
    v16h b = __builtin_shufflevector(b0, b1, 0, 1, 2, 3, 4, 5, 6, 7,
                                             8, 9, 10, 11, 12, 13, 14, 15);
    acc = __builtin_amdgcn_wmma_f32_16x16x32_f16(false, a, false, b, (short)0, acc,
                                                 false, false);
  }
  return acc;
}

// ---------------- kernel 1: f16 conversion + transposition of loop-invariant data ----
__global__ void prep_kernel(const float* __restrict__ enc, const float* __restrict__ Wa,
                            const float* __restrict__ Wr, const float* __restrict__ Ua,
                            _Float16* __restrict__ enc16, _Float16* __restrict__ WaT,
                            _Float16* __restrict__ WrT, _Float16* __restrict__ UaT) {
  const int i = blockIdx.x * blockDim.x + threadIdx.x;
  const int stride = gridDim.x * blockDim.x;
  for (int j = i; j < B_ * T_ * E_; j += stride) enc16[j] = (_Float16)enc[j];
  for (int j = i; j < T_ * 2 * D_; j += stride) {           // WaT[n][k] = Wa[k][n]
    int n = j / (2 * D_), k = j % (2 * D_);
    WaT[j] = (_Float16)Wa[k * T_ + n];
  }
  for (int j = i; j < 4 * D_ * D_; j += stride) {           // WrT[n][k] = Wr[k][n]
    int n = j / D_, k = j % D_;
    WrT[j] = (_Float16)Wr[k * (4 * D_) + n];
  }
  for (int j = i; j < T_ * E_; j += stride) {               // UaT[n][k] = Ua[k][n]
    int n = j / E_, k = j % E_;
    UaT[j] = (_Float16)Ua[k * T_ + n];
  }
}

// ---------------- kernel 2: init state + sync counters ----------------
__global__ void dec_init_kernel(const float* __restrict__ ih, const float* __restrict__ ic,
                                const float* __restrict__ ictx, float* __restrict__ h,
                                float* __restrict__ c, float* __restrict__ ctx,
                                _Float16* __restrict__ hc16, unsigned* __restrict__ sync) {
  int i = blockIdx.x * blockDim.x + threadIdx.x;
  if (i < B_ * D_) {
    h[i] = ih[i];
    c[i] = ic[i];
    int b = i / D_, d = i % D_;
    hc16[b * (2 * D_) + d]      = (_Float16)ih[i];
    hc16[b * (2 * D_) + D_ + d] = (_Float16)ic[i];
  }
  if (i < B_ * E_) ctx[i] = ictx[i];
  if (i < 2) sync[i] = 0u;
}

// ---------------- kernel 3: u_enc = enc @ Ua + bu, stored f16 (fits 192MB L2) ----
// M = B*T = 65536, N = T = 512, K = E = 128. One wave per 16x16 tile.
__global__ void __launch_bounds__(256) uenc_kernel(const _Float16* __restrict__ enc16,
                                                   const _Float16* __restrict__ UaT,
                                                   const float* __restrict__ bu,
                                                   _Float16* __restrict__ uenc) {
  const int lane = threadIdx.x & 31;
  const int wave = blockIdx.x * (blockDim.x >> 5) + (threadIdx.x >> 5);
  const int tiles_n = T_ / 16; // 32
  const int tm = wave / tiles_n;
  const int tn = wave % tiles_n;
  const int mlo = lane & 15, kh = lane >> 4;

  v8f acc = {0.f, 0.f, 0.f, 0.f, 0.f, 0.f, 0.f, 0.f};
  acc = wmma_tile(acc, enc16 + (size_t)tm * 16 * E_, E_,
                  UaT + (size_t)tn * 16 * E_, E_, E_, lane);

#pragma unroll
  for (int r = 0; r < 8; ++r) {
    int row = tm * 16 + r + kh * 8;
    int col = tn * 16 + mlo;
    uenc[(size_t)row * T_ + col] = (_Float16)(acc[r] + bu[col]);
  }
}

// ---------------- kernel 4: persistent scan over T steps ----------------
__global__ void __launch_bounds__(256, 1) decoder_kernel(
    const float* __restrict__ enc, const float* __restrict__ labels,
    const float* __restrict__ ba, const float* __restrict__ Va,
    const float* __restrict__ bv, const float* __restrict__ Wc,
    const float* __restrict__ bc, const float* __restrict__ Wk,
    const float* __restrict__ bl, const float* __restrict__ W1,
    const float* __restrict__ b1, const float* __restrict__ W2,
    const float* __restrict__ b2, const _Float16* __restrict__ uenc,
    const _Float16* __restrict__ WaT, const _Float16* __restrict__ WrT,
    float* __restrict__ s, float* __restrict__ sc, float* __restrict__ zrec,
    float* __restrict__ h, float* __restrict__ c, float* __restrict__ ctx,
    _Float16* __restrict__ hc16, unsigned* __restrict__ sync_cnt,
    unsigned* __restrict__ sync_gen, float* __restrict__ out) {
  const int tid  = threadIdx.x;
  const int lane = tid & 31;
  const int wg   = blockIdx.x * (blockDim.x >> 5) + (tid >> 5); // 0..2047
  const int mlo  = lane & 15, kh = lane >> 4;
  __shared__ float lds[512];

  const float bv0 = bv[0], bc0 = bc[0], b20 = b2[0];

  for (int t = 0; t < T_; ++t) {
    // ---- Phase A (WMMA): s = [h|c]@Wa + ba  and  zrec = h@Wr (independent) ----
    if (wg < 256) {
      // s tile: M=B(128) x N=T(512), K=2D(256); A = hc16 (lda = 2D)
      const int tm = wg >> 5, tn = wg & 31;
      v8f acc = {0.f, 0.f, 0.f, 0.f, 0.f, 0.f, 0.f, 0.f};
      acc = wmma_tile(acc, hc16 + (size_t)tm * 16 * (2 * D_), 2 * D_,
                      WaT + (size_t)tn * 16 * (2 * D_), 2 * D_, 2 * D_, lane);
#pragma unroll
      for (int r = 0; r < 8; ++r) {
        int row = tm * 16 + r + kh * 8;
        int col = tn * 16 + mlo;
        s[row * T_ + col] = acc[r] + ba[col];
      }
    } else if (wg < 512) {
      // zrec tile: M=B(128) x N=4D(512), K=D(128); A = h half of hc16 (lda = 2D)
      const int w2 = wg - 256;
      const int tm = w2 >> 5, tn = w2 & 31;
      v8f acc = {0.f, 0.f, 0.f, 0.f, 0.f, 0.f, 0.f, 0.f};
      acc = wmma_tile(acc, hc16 + (size_t)tm * 16 * (2 * D_), 2 * D_,
                      WrT + (size_t)tn * 16 * D_, D_, D_, lane);
#pragma unroll
      for (int r = 0; r < 8; ++r) {
        int row = tm * 16 + r + kh * 8;
        int col = tn * 16 + mlo;
        zrec[row * (4 * D_) + col] = acc[r];
      }
    }
    grid_sync(sync_cnt, sync_gen);

    // ---- Phase B (hw-tanh): score[b,tt] = bv + sum_n Va[n]*tanh(uenc + s) ----
    // 65536 rows over 2048 waves; lane owns 16 contiguous n -> 2x b128 f16 loads.
    for (int i = 0; i < 32; ++i) {
      const int row = wg * 32 + i;
      const int b = row >> 9, tt = row & (T_ - 1);
      const _Float16* up = uenc + (size_t)(b * T_ + tt) * T_ + lane * 16;
      const float* sp = s + b * T_ + lane * 16;
      const float* vp = Va + lane * 16;
      __builtin_prefetch(up + T_, 0, 0); // next row (global_prefetch_b8)
      v8h u0 = *(const v8h*)up;
      v8h u1 = *(const v8h*)(up + 8);
      float acc = 0.f;
#pragma unroll
      for (int q = 0; q < 8; ++q) acc += vp[q] * htanh((float)u0[q] + sp[q]);
#pragma unroll
      for (int q = 0; q < 8; ++q) acc += vp[8 + q] * htanh((float)u1[q] + sp[8 + q]);
#pragma unroll
      for (int o = 16; o > 0; o >>= 1) acc += __shfl_xor(acc, o, 32);
      if (lane == 0) sc[row] = acc + bv0;
    }
    grid_sync(sync_cnt, sync_gen);

    // ---- Phase C (one block per batch row): softmax, ctx, y, LSTM cell ----
    if (blockIdx.x < B_) {
      const int b = blockIdx.x;
      float* row = sc + b * T_;
      // max
      float m = -1e30f;
      for (int j = tid; j < T_; j += 256) m = fmaxf(m, row[j]);
      lds[tid] = m;
      __syncthreads();
      for (int o = 128; o > 0; o >>= 1) {
        if (tid < o) lds[tid] = fmaxf(lds[tid], lds[tid + o]);
        __syncthreads();
      }
      const float mx = lds[0];
      __syncthreads();
      // exp + sum (unnormalized weights kept in-place; normalize inside ctx)
      float sm = 0.f;
      for (int j = tid; j < T_; j += 256) {
        float e = __expf(row[j] - mx);
        row[j] = e;
        sm += e;
      }
      lds[tid] = sm;
      __syncthreads();
      for (int o = 128; o > 0; o >>= 1) {
        if (tid < o) lds[tid] += lds[tid + o];
        __syncthreads();
      }
      const float inv = 1.f / lds[0];
      __syncthreads();
      // ctx[b,e] = sum_t w * enc[b,t,e]   (threads 0..127 = e; coalesced over e)
      if (tid < E_) {
        float a = 0.f;
        const float* ep = enc + (size_t)b * T_ * E_ + tid;
        for (int j = 0; j < T_; ++j) a += row[j] * ep[(size_t)j * E_];
        a *= inv;
        ctx[b * E_ + tid] = a;
        lds[tid] = Wc[1 + tid] * a;
      }
      __syncthreads();
      for (int o = 64; o > 0; o >>= 1) {
        if (tid < o) lds[tid] += lds[tid + o];
        __syncthreads();
      }
      if (tid == 0) lds[0] = bc0 + Wc[0] * labels[b * T_ + t] + lds[0]; // y[b]
      __syncthreads();
      const float yb = lds[0];
      // Keras LSTM cell: gates i,f,g,o; f32 carry, f16 shadow for next GEMM
      if (tid < D_) {
        const int d = tid;
        const float* zp = zrec + b * 4 * D_;
        float gi = zp[d]          + yb * Wk[d]          + bl[d];
        float gf = zp[D_ + d]     + yb * Wk[D_ + d]     + bl[D_ + d];
        float gg = zp[2 * D_ + d] + yb * Wk[2 * D_ + d] + bl[2 * D_ + d];
        float go = zp[3 * D_ + d] + yb * Wk[3 * D_ + d] + bl[3 * D_ + d];
        float cn = sigf(gf) * c[b * D_ + d] + sigf(gi) * htanh(gg);
        float hn = sigf(go) * htanh(cn);
        c[b * D_ + d] = cn;
        h[b * D_ + d] = hn;
        hc16[b * (2 * D_) + d]      = (_Float16)hn;
        hc16[b * (2 * D_) + D_ + d] = (_Float16)cn;
      }
      __syncthreads();
    }
    grid_sync(sync_cnt, sync_gen); // h,c visible before next step's Phase A
  }

  // ---- Final: pred[b] = (concat(h,ctx) @ W1 + b1) @ W2 + b2 ----
  if (blockIdx.x < B_) {
    const int b = blockIdx.x;
    if (tid < D_) {
      float tmp = b1[tid];
      for (int j = 0; j < D_; ++j) tmp += h[b * D_ + j] * W1[j * D_ + tid];
      for (int j = 0; j < E_; ++j) tmp += ctx[b * E_ + j] * W1[(D_ + j) * D_ + tid];
      lds[tid] = tmp * W2[tid];
    }
    __syncthreads();
    for (int o = 64; o > 0; o >>= 1) {
      if (tid < o) lds[tid] += lds[tid + o];
      __syncthreads();
    }
    if (tid == 0) out[b] = lds[0] + b20;
  }
}

// ---------------- host launcher ----------------
extern "C" void kernel_launch(void* const* d_in, const int* in_sizes, int n_in,
                              void* d_out, int out_size, void* d_ws, size_t ws_size,
                              hipStream_t stream) {
  const float* enc    = (const float*)d_in[0];
  const float* labels = (const float*)d_in[1];
  const float* ih     = (const float*)d_in[2];
  const float* ic     = (const float*)d_in[3];
  const float* ictx   = (const float*)d_in[4];
  const float* Wa     = (const float*)d_in[5];
  const float* ba     = (const float*)d_in[6];
  const float* Ua     = (const float*)d_in[7];
  const float* bu     = (const float*)d_in[8];
  const float* Va     = (const float*)d_in[9];
  const float* bv     = (const float*)d_in[10];
  const float* Wc     = (const float*)d_in[11];
  const float* bc     = (const float*)d_in[12];
  const float* Wk     = (const float*)d_in[13];
  const float* Wr     = (const float*)d_in[14];
  const float* bl     = (const float*)d_in[15];
  const float* W1     = (const float*)d_in[16];
  const float* b1     = (const float*)d_in[17];
  const float* W2     = (const float*)d_in[18];
  const float* b2     = (const float*)d_in[19];

  char* ws = (char*)d_ws;
  _Float16* uenc  = (_Float16*)(ws + OFF_UENC);
  float* s        = (float*)(ws + OFF_S);
  float* sc       = (float*)(ws + OFF_SC);
  float* zrec     = (float*)(ws + OFF_Z);
  float* h        = (float*)(ws + OFF_H);
  float* c        = (float*)(ws + OFF_C);
  float* ctx      = (float*)(ws + OFF_CTX);
  unsigned* sync  = (unsigned*)(ws + OFF_SYNC);
  _Float16* enc16 = (_Float16*)(ws + OFF_ENC16);
  _Float16* WaT   = (_Float16*)(ws + OFF_WAT);
  _Float16* WrT   = (_Float16*)(ws + OFF_WRT);
  _Float16* UaT   = (_Float16*)(ws + OFF_UAT);
  _Float16* hc16  = (_Float16*)(ws + OFF_HC16);

  // 1) f16 conversion + weight transposition (once per call)
  prep_kernel<<<512, 256, 0, stream>>>(enc, Wa, Wr, Ua, enc16, WaT, WrT, UaT);

  // 2) init state + barrier counters (deterministic each call)
  dec_init_kernel<<<64, 256, 0, stream>>>(ih, ic, ictx, h, c, ctx, hc16, sync);

  // 3) WMMA precompute: u_enc (f16, +bu folded). (65536/16)x(512/16) tiles,
  //    8 waves per 256-thread block -> 16384 blocks.
  uenc_kernel<<<(B_ * T_ / 16) * (T_ / 16) / 8, 256, 0, stream>>>(enc16, UaT, bu, uenc);

  // 4) persistent scan: 256 blocks x 256 threads (2048 wave32s), grid-sync'd phases.
  decoder_kernel<<<256, 256, 0, stream>>>(enc, labels, ba, Va, bv, Wc, bc, Wk, bl,
                                          W1, b1, W2, b2, uenc, WaT, WrT, s, sc, zrec,
                                          h, c, ctx, hc16, sync, sync + 1,
                                          (float*)d_out);
}
